// Model_42734924595608
// MI455X (gfx1250) — matrix operand order repaired and checked by reading, not compiled, not run on hardware
//
#include <hip/hip_runtime.h>
#include <hip/hip_bf16.h>

typedef __bf16 bf16;
typedef __attribute__((ext_vector_type(16))) __bf16 v16bf;
typedef __attribute__((ext_vector_type(8)))  float  v8f;

// Pipelining pin: no-op asm that (a) cannot sink below the WMMA because the
// WMMA reads `a` which this asm defines, and (b) cannot hoist above the
// prefetch ds_loads because of the memory clobber. Guarantees
// prefetch-loads < pin < wmma, so waitcnt emits partial s_wait_dscnt.
#define SCHED_PIN(a) asm volatile("" : "+v"(a) :: "memory")

#define N_DIM 32
#define C_IN  64
#define COUT  64
#define T_DIM 300
#define V_DIM 25
#define KN    5
#define TV    (T_DIM * V_DIM)        // 7500
#define CNT_F 240000.0f              // N*T*V per-channel BN count
#define EPSBN 1e-5f

// ---- K1 (fused 1x1 conv + graph conv) tiling ----
#define TT1    8
#define COLS1  (TT1 * V_DIM)         // 200
#define COLS1P 208                   // 13 tiles of 16
#define KTOT   320                   // 5 * 64 contracted
#define NK1    (KTOT / 32)           // 10 WMMA K-steps
#define TB1    ((T_DIM + TT1 - 1) / TT1)   // 38

// ---- K3 (temporal conv) tiling ----
#define TT2    16
#define COLS2  (TT2 * V_DIM)         // 400 -> 25 tiles of 16
#define TIN    (TT2 + 8)             // 24 timesteps incl. halo 4+4
#define ICOLS  (TIN * V_DIM)         // 600
#define NK3    18                    // 9 taps * 2 K-steps of 32
#define TB2    ((T_DIM + TT2 - 1) / TT2)   // 19

// stats layout (floats): [0:64) bn1 sum, [64:128) bn1 sq, [128:192) bn1 scale,
// [192:256) bn1 shift, [256:320) bn2 sum, [320:384) bn2 sq, [384:448) bn2 scale,
// [448:512) bn2 shift
#define G_ELEMS 15360000

__global__ __launch_bounds__(512) void k0_zero(float* stats) {
    if (threadIdx.x < 512) stats[threadIdx.x] = 0.0f;
}

// =====================================================================
// K1: g = (W * xA) + bias-term ; accumulate BN1 sum/sumsq
// =====================================================================
__global__ __launch_bounds__(256) void k1_gcn(
    const float* __restrict__ x, const float* __restrict__ A,
    const float* __restrict__ Badp, const float* __restrict__ lamda,
    const float* __restrict__ conv_w, const float* __restrict__ conv_b,
    float* __restrict__ g, float* __restrict__ stats)
{
    __shared__ float adj[KN][V_DIM][V_DIM];   // adaptive part pre-scaled by lambda
    __shared__ float xt[C_IN][COLS1];         // fp32 x tile (async-filled)
    __shared__ bf16  w2[COUT][KTOT];          // w2[c][k*64+ci]
    __shared__ bf16  xa[COLS1P][KTOT];        // col-major for B-operand loads
    __shared__ float gbias[COUT][V_DIM];
    __shared__ float s_sum[COUT], s_sq[COUT];

    const int tid = threadIdx.x;
    const int n  = blockIdx.x / TB1;
    const int tb = blockIdx.x % TB1;
    const int t0 = tb * TT1;
    const int tv = min(TT1, T_DIM - t0);      // valid timesteps (8 or 4)
    const int validCols = tv * V_DIM;         // 200 or 100 (both %4 == 0)
    const float lam = lamda[0];

    // adjacency (fixed + lambda-scaled adaptive)
    for (int e = tid; e < KN * V_DIM * V_DIM; e += 256) {
        int k = e / (V_DIM * V_DIM), r = e % (V_DIM * V_DIM);
        float v;
        if (k < KN - 2) v = A[k * V_DIM * V_DIM + r];
        else            v = lam * Badp[((n * 2) + (k - (KN - 2))) * V_DIM * V_DIM + r];
        adj[k][r / V_DIM][r % V_DIM] = v;
    }
    // x tile: async global->LDS (128b per lane), bypassing VGPRs (ASYNCcnt path)
    for (int e = tid; e < C_IN * (COLS1 / 4); e += 256) {
        int ci = e / (COLS1 / 4), c4 = (e % (COLS1 / 4)) * 4;
        unsigned lds = (unsigned)(size_t)&xt[ci][c4];
        if (c4 < validCols) {
            const float* gp = &x[(size_t)(n * C_IN + ci) * TV + t0 * V_DIM + c4];
            asm volatile("global_load_async_to_lds_b128 %0, %1, off"
                         :: "v"(lds), "v"(gp) : "memory");
        } else {
            xt[ci][c4] = 0.0f; xt[ci][c4 + 1] = 0.0f;
            xt[ci][c4 + 2] = 0.0f; xt[ci][c4 + 3] = 0.0f;
        }
    }
    // weights -> bf16, laid out [c][k*64+ci]
    for (int e = tid; e < COUT * KTOT; e += 256) {
        int c = e / KTOT, kk = e % KTOT;
        int k = kk >> 6, ci = kk & 63;
        w2[c][kk] = (bf16)conv_w[((k * COUT + c) * C_IN) + ci];
    }
    if (tid < COUT) { s_sum[tid] = 0.0f; s_sq[tid] = 0.0f; }
    asm volatile("s_wait_asynccnt 0x0" ::: "memory");
    __syncthreads();

    // xA[k,ci,(t,w)] = sum_v x[ci,t,v] * adj[k][v][w]   (K=25 VALU contraction)
    for (int e = tid; e < COLS1P * KTOT; e += 256) {
        int col = e / KTOT, kk = e % KTOT;
        float acc = 0.0f;
        if (col < COLS1) {
            int k = kk >> 6, ci = kk & 63;
            int tl = col / V_DIM, w = col % V_DIM;
            #pragma unroll
            for (int v = 0; v < V_DIM; ++v)
                acc += xt[ci][tl * V_DIM + v] * adj[k][v][w];
        }
        xa[col][kk] = (bf16)acc;
    }
    // bias term: gbias[c][w] = sum_k b[k,c] * colsum_k[w]
    for (int e = tid; e < COUT * V_DIM; e += 256) {
        int c = e / V_DIM, w = e % V_DIM;
        float acc = 0.0f;
        for (int k = 0; k < KN; ++k) {
            float s = 0.0f;
            #pragma unroll
            for (int v = 0; v < V_DIM; ++v) s += adj[k][v][w];
            acc += conv_b[k * COUT + c] * s;
        }
        gbias[c][w] = acc;
    }
    __syncthreads();

    // WMMA GEMM: M=64 (4 tiles) x N=208 (13 tiles), K=320.
    // Wave owns a fixed M-tile -> all 10 A fragments live in registers.
    const int wid = tid >> 5, lane = tid & 31;
    const int half = lane >> 4, l15 = lane & 15;
    const int mt = wid & 3;
    const int m = mt * 16 + l15;

    v16bf afr[NK1];
    #pragma unroll
    for (int ki = 0; ki < NK1; ++ki) {
        int kb = ki * 32;
        #pragma unroll
        for (int j = 0; j < 4; ++j) {
            int k0 = kb + 2 * j + 8 * half;
            afr[ki][2 * j]     = w2[m][k0];
            afr[ki][2 * j + 1] = w2[m][k0 + 1];
            int k1 = kb + 16 + 2 * j + 8 * half;
            afr[ki][2 * (j + 4)]     = w2[m][k1];
            afr[ki][2 * (j + 4) + 1] = w2[m][k1 + 1];
        }
    }

    for (int nt = (wid >> 2); nt < 13; nt += 2) {
        const int col = nt * 16 + l15;
        auto loadB = [&](int ki, v16bf& b) {
            int kb = ki * 32;
            #pragma unroll
            for (int j = 0; j < 8; ++j) {
                int k0 = kb + 2 * j + 8 * half + ((j >> 2) << 3);
                b[2 * j]     = xa[col][k0];
                b[2 * j + 1] = xa[col][k0 + 1];
            }
        };
        v8f acc = {};
        v16bf b0, b1;                      // software-pipelined B fragments
        loadB(0, b0);
        #pragma unroll
        for (int ki = 0; ki < NK1; ++ki) {
            if (ki + 1 < NK1) loadB(ki + 1, (ki & 1) ? b0 : b1);
            SCHED_PIN(acc);                // prefetch-loads < pin < wmma
            acc = __builtin_amdgcn_wmma_f32_16x16x32_bf16(
                false, afr[ki], false, (ki & 1) ? b1 : b0, (short)0, acc,
                false, false);
        }
        // store + BN1 partial stats
        const int tl = col / V_DIM, wv = col % V_DIM;
        const bool valid = col < validCols;
        #pragma unroll
        for (int r = 0; r < 8; ++r) {
            int c = mt * 16 + r + 8 * half;
            float val = acc[r] + gbias[c][wv];
            if (valid) {
                g[(size_t)(n * COUT + c) * TV + (t0 + tl) * V_DIM + wv] = val;
                atomicAdd(&s_sum[c], val);
                atomicAdd(&s_sq[c], val * val);
            }
        }
    }
    __syncthreads();
    if (tid < COUT) {
        atomicAdd(&stats[tid], s_sum[tid]);
        atomicAdd(&stats[COUT + tid], s_sq[tid]);
    }
}

// =====================================================================
// K2/K4: finalize BN scale/shift from accumulated sum/sumsq
// =====================================================================
__global__ __launch_bounds__(64) void k2_bn_finalize(
    const float* __restrict__ acc, const float* __restrict__ gamma,
    const float* __restrict__ beta, float* __restrict__ scale,
    float* __restrict__ shift)
{
    int c = threadIdx.x;
    float mean = acc[c] / CNT_F;
    float var  = acc[COUT + c] / CNT_F - mean * mean;
    float sc = gamma[c] * rsqrtf(var + EPSBN);
    scale[c] = sc;
    shift[c] = beta[c] - mean * sc;
}

// =====================================================================
// K3: h = tcn(relu(bn1(g))) + bias ; accumulate BN2 sum/sumsq
// =====================================================================
__global__ __launch_bounds__(256) void k3_tcn(
    const float* __restrict__ g, const float* __restrict__ tcn_w,
    const float* __restrict__ tcn_b, const float* __restrict__ stats,
    float* __restrict__ h)
{
    __shared__ bf16  hin[ICOLS][COUT];    // bn1+relu'd input, col-major
    __shared__ bf16  wt[9][COUT][COUT];   // wt[kt][co][ci]
    __shared__ float sc1[COUT], sh1[COUT];
    __shared__ float s_sum[COUT], s_sq[COUT];

    const int tid = threadIdx.x;
    const int n  = blockIdx.x / TB2;
    const int tb = blockIdx.x % TB2;
    const int t0 = tb * TT2;

    if (tid < COUT) {
        sc1[tid] = stats[128 + tid];
        sh1[tid] = stats[192 + tid];
        s_sum[tid] = 0.0f; s_sq[tid] = 0.0f;
    }
    __syncthreads();

    for (int e = tid; e < 9 * COUT * COUT; e += 256) {
        int kt = e / (COUT * COUT), r = e % (COUT * COUT);
        int co = r >> 6, ci = r & 63;
        wt[kt][co][ci] = (bf16)tcn_w[(size_t)(co * COUT + ci) * 9 + kt];
    }
    for (int e = tid; e < C_IN * ICOLS; e += 256) {
        int ci = e / ICOLS, icol = e % ICOLS;
        int base = (t0 - 4) * V_DIM + icol;      // == tin*25 + v
        float val = 0.0f;
        if (base >= 0 && base < TV)
            val = g[(size_t)(n * C_IN + ci) * TV + base];
        val = fmaxf(val * sc1[ci] + sh1[ci], 0.0f);
        hin[icol][ci] = (bf16)val;
    }
    __syncthreads();

    // WMMA: M=64 (4 tiles) x N=400 (25 tiles), K = 9 taps * 64 ch.
    // Wave owns a fixed M-tile -> all 18 A fragments hoisted to registers.
    const int wid = tid >> 5, lane = tid & 31;
    const int half = lane >> 4, l15 = lane & 15;
    const int mt = wid & 3;
    const int m = mt * 16 + l15;

    v16bf afr[NK3];
    #pragma unroll
    for (int ki = 0; ki < NK3; ++ki) {
        int kt = ki >> 1, kb = (ki & 1) * 32;
        #pragma unroll
        for (int j = 0; j < 4; ++j) {
            int k0 = kb + 2 * j + 8 * half;
            afr[ki][2 * j]     = wt[kt][m][k0];
            afr[ki][2 * j + 1] = wt[kt][m][k0 + 1];
            int k1 = kb + 16 + 2 * j + 8 * half;
            afr[ki][2 * (j + 4)]     = wt[kt][m][k1];
            afr[ki][2 * (j + 4) + 1] = wt[kt][m][k1 + 1];
        }
    }

    for (int nt = (wid >> 2); nt < 25; nt += 2) {
        const int colo = nt * 16 + l15;
        auto loadB = [&](int ki, v16bf& b) {
            int kt = ki >> 1, kb = (ki & 1) * 32;
            int icol = colo + kt * V_DIM;         // temporally shifted column
            #pragma unroll
            for (int j = 0; j < 8; ++j) {
                int k0 = kb + 2 * j + 8 * half + ((j >> 2) << 3);
                b[2 * j]     = hin[icol][k0];
                b[2 * j + 1] = hin[icol][k0 + 1];
            }
        };
        v8f acc = {};
        v16bf bA0, bA1, bB0, bB1;          // ping-pong chunk pairs (per kt)
        loadB(0, bA0); loadB(1, bA1);
        #pragma unroll
        for (int kt = 0; kt < 9; ++kt) {
            const bool odd = (kt & 1) != 0;
            if (kt + 1 < 9) {
                loadB(2 * (kt + 1),     odd ? bA0 : bB0);
                loadB(2 * (kt + 1) + 1, odd ? bA1 : bB1);
            }
            SCHED_PIN(acc);                // prefetch-loads < pin < wmma
            acc = __builtin_amdgcn_wmma_f32_16x16x32_bf16(
                false, afr[2 * kt], false, odd ? bB0 : bA0, (short)0, acc,
                false, false);
            acc = __builtin_amdgcn_wmma_f32_16x16x32_bf16(
                false, afr[2 * kt + 1], false, odd ? bB1 : bA1, (short)0, acc,
                false, false);
        }
        const bool valid = (t0 * V_DIM + colo) < TV;
        #pragma unroll
        for (int r = 0; r < 8; ++r) {
            int co = mt * 16 + r + 8 * half;
            float val = acc[r] + tcn_b[co];
            if (valid) {
                h[(size_t)(n * COUT + co) * TV + t0 * V_DIM + colo] = val;
                atomicAdd(&s_sum[co], val);
                atomicAdd(&s_sq[co], val * val);
            }
        }
    }
    __syncthreads();
    if (tid < COUT) {
        atomicAdd((float*)&stats[256 + tid], s_sum[tid]);
        atomicAdd((float*)&stats[320 + tid], s_sq[tid]);
    }
}

// =====================================================================
// K5: out = relu(bn2(h) + x)   (vectorized float4; 7500 % 4 == 0)
// =====================================================================
__global__ __launch_bounds__(256) void k5_epilogue(
    const float* __restrict__ h, const float* __restrict__ x,
    const float* __restrict__ stats, float* __restrict__ out)
{
    size_t e4 = (size_t)blockIdx.x * blockDim.x + threadIdx.x;
    if (e4 >= G_ELEMS / 4) return;
    size_t e = e4 * 4;
    int c = (int)((e / TV) % COUT);
    float sc = stats[384 + c], sh = stats[448 + c];
    float4 hv = ((const float4*)h)[e4];
    float4 xv = ((const float4*)x)[e4];
    float4 o;
    o.x = fmaxf(hv.x * sc + sh + xv.x, 0.0f);
    o.y = fmaxf(hv.y * sc + sh + xv.y, 0.0f);
    o.z = fmaxf(hv.z * sc + sh + xv.z, 0.0f);
    o.w = fmaxf(hv.w * sc + sh + xv.w, 0.0f);
    ((float4*)out)[e4] = o;
}

extern "C" void kernel_launch(void* const* d_in, const int* in_sizes, int n_in,
                              void* d_out, int out_size, void* d_ws, size_t ws_size,
                              hipStream_t stream) {
    const float* x      = (const float*)d_in[0];
    const float* A      = (const float*)d_in[1];
    const float* Badp   = (const float*)d_in[2];
    const float* lamda  = (const float*)d_in[3];
    const float* conv_w = (const float*)d_in[4];
    const float* conv_b = (const float*)d_in[5];
    const float* bn1_g  = (const float*)d_in[6];
    const float* bn1_b  = (const float*)d_in[7];
    const float* tcn_w  = (const float*)d_in[8];
    const float* tcn_b  = (const float*)d_in[9];
    const float* bn2_g  = (const float*)d_in[10];
    const float* bn2_b  = (const float*)d_in[11];
    float* out = (float*)d_out;

    float* g     = (float*)d_ws;
    float* h     = g + G_ELEMS;
    float* stats = h + G_ELEMS;

    k0_zero<<<1, 512, 0, stream>>>(stats);
    k1_gcn<<<N_DIM * TB1, 256, 0, stream>>>(x, A, Badp, lamda, conv_w, conv_b, g, stats);
    k2_bn_finalize<<<1, 64, 0, stream>>>(stats, bn1_g, bn1_b, stats + 128, stats + 192);
    k3_tcn<<<N_DIM * TB2, 256, 0, stream>>>(g, tcn_w, tcn_b, stats, h);
    k2_bn_finalize<<<1, 64, 0, stream>>>(stats + 256, bn2_g, bn2_b, stats + 384, stats + 448);
    k5_epilogue<<<(G_ELEMS / 4 + 255) / 256, 256, 0, stream>>>(h, x, stats, out);
    (void)in_sizes; (void)n_in; (void)out_size; (void)ws_size;
}
